// BaseEmbedder_14448269984433
// MI455X (gfx1250) — matrix operand order repaired and checked by reading, not compiled, not run on hardware
//
#include <hip/hip_runtime.h>

// ---------------------------------------------------------------------------
// KNN (k=5) + inverse-distance-weighted aux aggregation, fused with a
// WMMA-f16 split-precision distance GEMM for MI455X (gfx1250, wave32).
//
//   dist^2(q,db) = |q|^2 + |db|^2 - 2*(q . db)
//   q.db computed as (qh+ql).(dh+dl) ~= qh.dh + qh.dl + ql.dh  (3x v_wmma
//   f32_16x16x32_f16, K=32 == feature dim -> one K-step per tile).
//
// Workspace layout (d_ws, ~9.2 MB needed):
//   [0        ) db_hi  : 65536*32 f16   (4,194,304 B)
//   [4194304  ) db_lo  : 65536*32 f16   (4,194,304 B)
//   [8388608  ) q_hi   :  4096*32 f16   (  262,144 B)
//   [8650752  ) q_lo   :  4096*32 f16   (  262,144 B)
//   [8912896  ) db_sq  : 65536    f32   (  262,144 B)
//   [9175040  ) q_sq   :  4096    f32   (   16,384 B)
// ---------------------------------------------------------------------------

#define N_Q    4096
#define N_DB   65536
#define DIM    32
#define KNN    5
#define EPSW   1e-6f
#define WAVES  4
#define TILE_Q 16

typedef __attribute__((ext_vector_type(16))) _Float16     v16h;
typedef __attribute__((ext_vector_type(8)))  float        v8f;
typedef __attribute__((ext_vector_type(4)))  unsigned int v4u;

union Frag { v16h h; v4u u[2]; };

// ---------------------------------------------------------------------------
// Prep: f32 -> (hi,lo) f16 split + squared norms. One thread per row.
// ---------------------------------------------------------------------------
__global__ __launch_bounds__(256) void knn_prep(
    const float* __restrict__ q, const float* __restrict__ db,
    _Float16* __restrict__ qh, _Float16* __restrict__ ql,
    _Float16* __restrict__ dbh, _Float16* __restrict__ dbl,
    float* __restrict__ qsq, float* __restrict__ dbsq)
{
  int i = blockIdx.x * blockDim.x + threadIdx.x;
  const float* src; _Float16 *h, *l; float* sq; int row;
  if (i < N_DB)            { src = db; h = dbh; l = dbl; sq = dbsq; row = i; }
  else if (i < N_DB + N_Q) { src = q;  h = qh;  l = ql;  sq = qsq;  row = i - N_DB; }
  else return;

  const float* p = src + (size_t)row * DIM;
  float acc = 0.f;
#pragma unroll
  for (int k = 0; k < DIM; ++k) {
    float x = p[k];
    acc = fmaf(x, x, acc);
    _Float16 hi = (_Float16)x;
    _Float16 lo = (_Float16)(x - (float)hi);
    h[(size_t)row * DIM + k] = hi;
    l[(size_t)row * DIM + k] = lo;
  }
  sq[row] = acc;
}

// ---------------------------------------------------------------------------
// Main: each block = one 16-query tile; 4 waves each scan 1/4 of the DB,
// keeping per-lane in-register sorted top-5 per accumulator row; merge via
// LDS; then gather + weighted-sum the aux rows.
// ---------------------------------------------------------------------------
__global__ __launch_bounds__(128) void knn_main(
    const _Float16* __restrict__ qh,  const _Float16* __restrict__ ql,
    const _Float16* __restrict__ dbh, const _Float16* __restrict__ dbl,
    const float* __restrict__ qsq,    const float* __restrict__ dbsq,
    const float* __restrict__ aux,    float* __restrict__ out)
{
  __shared__ float sd[TILE_Q][WAVES * 16][KNN];
  __shared__ int   si[TILE_Q][WAVES * 16][KNN];
  __shared__ float fw[TILE_Q][KNN];
  __shared__ int   fi[TILE_Q][KNN];

  const int tid   = threadIdx.x;
  const int wave  = tid >> 5;
  const int lane  = tid & 31;
  const int half  = lane >> 4;   // 0: lanes 0-15, 1: lanes 16-31
  const int l16   = lane & 15;
  const int qbase = blockIdx.x * TILE_Q;

  // ---- A fragments (16x32 f16): lane holds row M=l16,
  //      K in [8h,8h+8) U [16+8h,16+8h+8)  ->  two 16-byte chunks of the row.
  const v4u* qrh = (const v4u*)(qh + (size_t)(qbase + l16) * DIM);
  const v4u* qrl = (const v4u*)(ql + (size_t)(qbase + l16) * DIM);
  Frag a_hi, a_lo;
  a_hi.u[0] = qrh[half]; a_hi.u[1] = qrh[2 + half];
  a_lo.u[0] = qrl[half]; a_lo.u[1] = qrl[2 + half];

  // C layout: vgpr r holds (M = r + 8*half, N = n_base + l16)
  float qs[8];
#pragma unroll
  for (int r = 0; r < 8; ++r) qs[r] = qsq[qbase + r + 8 * half];

  float td[8][KNN];
  int   ti[8][KNN];
#pragma unroll
  for (int r = 0; r < 8; ++r)
#pragma unroll
    for (int j = 0; j < KNN; ++j) { td[r][j] = 3.0e38f; ti[r][j] = 0; }

  const int nBeg = wave * (N_DB / WAVES);
  const int nEnd = nBeg + (N_DB / WAVES);

  for (int n = nBeg; n < nEnd; n += 16) {
    const int col = n + l16;
    // ---- B fragments (32x16 f16): lane holds column N=l16,
    //      K in [8h,8h+8) U [16+8h,16+8h+8)  ->  two 16-byte chunks of the DB row.
    const v4u* brh = (const v4u*)(dbh + (size_t)col * DIM);
    const v4u* brl = (const v4u*)(dbl + (size_t)col * DIM);
    Frag b_hi, b_lo;
    b_hi.u[0] = brh[half]; b_hi.u[1] = brh[2 + half];
    b_lo.u[0] = brl[half]; b_lo.u[1] = brl[2 + half];
    const float dsq = dbsq[col];

    if (n + 128 < nEnd) {  // stream-ahead into caches
      __builtin_prefetch(dbh + (size_t)(col + 128) * DIM, 0, 0);
      __builtin_prefetch(dbl + (size_t)(col + 128) * DIM, 0, 0);
    }

    v8f c = {};
    c = __builtin_amdgcn_wmma_f32_16x16x32_f16(false, a_hi.h, false, b_hi.h,
                                               (short)0, c, false, false);
    c = __builtin_amdgcn_wmma_f32_16x16x32_f16(false, a_hi.h, false, b_lo.h,
                                               (short)0, c, false, false);
    c = __builtin_amdgcn_wmma_f32_16x16x32_f16(false, a_lo.h, false, b_hi.h,
                                               (short)0, c, false, false);

#pragma unroll
    for (int r = 0; r < 8; ++r) {
      float x = fmaf(-2.0f, c[r], qs[r] + dsq);     // dist^2
      if (x < td[r][KNN - 1]) {                     // wave-coherent skip
        td[r][KNN - 1] = x; ti[r][KNN - 1] = col;
#pragma unroll
        for (int j = KNN - 2; j >= 0; --j) {
          if (td[r][j + 1] < td[r][j]) {
            float tf = td[r][j]; td[r][j] = td[r][j + 1]; td[r][j + 1] = tf;
            int   tn = ti[r][j]; ti[r][j] = ti[r][j + 1]; ti[r][j + 1] = tn;
          }
        }
      }
    }
  }

  // ---- spill per-lane top-k candidates to LDS
#pragma unroll
  for (int r = 0; r < 8; ++r) {
    const int row  = r + 8 * half;
    const int slot = wave * 16 + l16;
#pragma unroll
    for (int j = 0; j < KNN; ++j) {
      sd[row][slot][j] = td[r][j];
      si[row][slot][j] = ti[r][j];
    }
  }
  __syncthreads();

  // ---- per-row merge of 64 sorted 5-lists + weight computation
  if (tid < TILE_Q) {
    float bd[KNN]; int bi[KNN];
#pragma unroll
    for (int j = 0; j < KNN; ++j) { bd[j] = 3.0e38f; bi[j] = 0; }
    for (int s = 0; s < WAVES * 16; ++s) {
#pragma unroll
      for (int j = 0; j < KNN; ++j) {
        float x = sd[tid][s][j];
        if (x >= bd[KNN - 1]) break;   // slot list is sorted ascending
        int ix = si[tid][s][j];
        bd[KNN - 1] = x; bi[KNN - 1] = ix;
#pragma unroll
        for (int m = KNN - 2; m >= 0; --m) {
          if (bd[m + 1] < bd[m]) {
            float tf = bd[m]; bd[m] = bd[m + 1]; bd[m + 1] = tf;
            int   tn = bi[m]; bi[m] = bi[m + 1]; bi[m + 1] = tn;
          }
        }
      }
    }
    float w[KNN], wsum = 0.f;
#pragma unroll
    for (int j = 0; j < KNN; ++j) {
      w[j] = 1.0f / (sqrtf(fmaxf(bd[j], 0.0f)) + EPSW);
      wsum += w[j];
    }
    const float inv = 1.0f / wsum;
#pragma unroll
    for (int j = 0; j < KNN; ++j) { fw[tid][j] = w[j] * inv; fi[tid][j] = bi[j]; }
  }
  __syncthreads();

  // ---- gather aux rows and write weighted sum (whole block, coalesced)
  const int colOut = tid & 31;
  for (int rr = tid >> 5; rr < TILE_Q; rr += WAVES) {
    float acc = 0.f;
#pragma unroll
    for (int j = 0; j < KNN; ++j)
      acc = fmaf(fw[rr][j], aux[(size_t)fi[rr][j] * DIM + colOut], acc);
    out[(size_t)(qbase + rr) * DIM + colOut] = acc;
  }
}

// ---------------------------------------------------------------------------
extern "C" void kernel_launch(void* const* d_in, const int* in_sizes, int n_in,
                              void* d_out, int out_size, void* d_ws, size_t ws_size,
                              hipStream_t stream) {
  (void)in_sizes; (void)n_in; (void)out_size; (void)ws_size;
  const float* q   = (const float*)d_in[0];   // [4096, 32]
  const float* db  = (const float*)d_in[1];   // [65536, 32]
  const float* aux = (const float*)d_in[2];   // [65536, 32]
  float* out = (float*)d_out;                 // [4096, 32]

  char* ws = (char*)d_ws;
  _Float16* dbh  = (_Float16*)(ws + 0);
  _Float16* dbl  = (_Float16*)(ws + 4194304);
  _Float16* qhp  = (_Float16*)(ws + 8388608);
  _Float16* qlp  = (_Float16*)(ws + 8650752);
  float*    dbsq = (float*)   (ws + 8912896);
  float*    qsq  = (float*)   (ws + 9175040);

  knn_prep<<<(N_DB + N_Q + 255) / 256, 256, 0, stream>>>(
      q, db, qhp, qlp, dbh, dbl, qsq, dbsq);
  knn_main<<<N_Q / TILE_Q, WAVES * 32, 0, stream>>>(
      qhp, qlp, dbh, dbl, qsq, dbsq, aux, out);
}
